// MuonOptimizerFixed_73916387164392
// MI455X (gfx1250) — compile-verified, matrix operand order checked by reading, not run on hardware
//
#include <hip/hip_runtime.h>
#include <hip/hip_bf16.h>

// ---------------------------------------------------------------------------
// MI455X (gfx1250) multi-head attention forward.
//   bf16 WMMA, double-buffered TDM (tensor_load_to_lds) tile staging,
//   flash softmax with DPP16 row-max and WMMA row-sum.
//   D_MODEL=1024, H=16, DK=64, B=4, S=2048
// ---------------------------------------------------------------------------

typedef __attribute__((ext_vector_type(16))) __bf16 bf16x16;
typedef __attribute__((ext_vector_type(8)))  __bf16 bf16x8;
typedef __attribute__((ext_vector_type(8)))  float  f32x8;
typedef __attribute__((ext_vector_type(4)))  float  f32x4;
typedef __attribute__((ext_vector_type(4)))  unsigned int u32x4;
typedef __attribute__((ext_vector_type(8)))  int    i32x8;
typedef __attribute__((ext_vector_type(4)))  int    i32x4;

#define D_MODEL 1024
#define NHEADS  16
#define DK      64
#define BATCH   4
#define SEQ     2048

#if __has_builtin(__builtin_amdgcn_tensor_load_to_lds)
#define HAVE_TDM 1
#endif

// ---------------------------------------------------------------------------
// TDM 2-D tile load: global (row-major, stride0 elems of 2B) -> LDS.
// Optional LDS padding: after every 16 DWORDs, skip 8 DWORDs
// (turns a 32-elem bf16 row into a 48-elem LDS stride).
// ---------------------------------------------------------------------------
__device__ __forceinline__ void tdm_load_2d(unsigned lds_addr, const void* gptr,
                                            unsigned tensor_d0, unsigned tensor_d1,
                                            unsigned tile_d0, unsigned tile_d1,
                                            unsigned long long stride0,
                                            bool pad48)
{
#if defined(HAVE_TDM)
    const unsigned long long ga = (unsigned long long)(uintptr_t)gptr;
    u32x4 g0;
    g0[0] = 1u;                                        // count=1, user descriptor
    g0[1] = lds_addr;                                  // LDS byte address
    g0[2] = (unsigned)(ga & 0xffffffffull);            // global_addr[31:0]
    g0[3] = (unsigned)((ga >> 32) & 0x01ffffffull) | (2u << 30);   // addr[56:32] | type=2
    i32x8 g1;
    unsigned w0 = (1u << 16);                          // data_size=1 (2 bytes), mask=0
    if (pad48) w0 |= (1u << 20) | (3u << 22) | (7u << 25); // pad_en, every 16 dw, +8 dw
    g1[0] = (int)w0;
    g1[1] = (int)((tensor_d0 & 0xffffu) << 16);        // atomic_addr=0 | dim0[15:0]
    g1[2] = (int)(((tensor_d0 >> 16) & 0xffffu) | ((tensor_d1 & 0xffffu) << 16));
    g1[3] = (int)(((tensor_d1 >> 16) & 0xffffu) | ((tile_d0 & 0xffffu) << 16));
    g1[4] = (int)(tile_d1 & 0xffffu);                  // tile_dim1 | tile_dim2=0
    g1[5] = (int)(stride0 & 0xffffffffull);            // dim0_stride[31:0]
    g1[6] = (int)((stride0 >> 32) & 0xffffull);        // dim0_stride hi | dim1_stride lo
    g1[7] = 0;
    const i32x4 z4 = {0, 0, 0, 0};
#if __clang_major__ >= 23
    const i32x8 z8 = {0, 0, 0, 0, 0, 0, 0, 0};
    __builtin_amdgcn_tensor_load_to_lds(g0, g1, z4, z4, z8, 0);
#else
    __builtin_amdgcn_tensor_load_to_lds(g0, g1, z4, z4, 0);
#endif
#else
    (void)lds_addr; (void)gptr; (void)tensor_d0; (void)tensor_d1;
    (void)tile_d0; (void)tile_d1; (void)stride0; (void)pad48;
#endif
}

__device__ __forceinline__ float fast_exp2(float x)
{
#if __has_builtin(__builtin_amdgcn_exp2f)
    return __builtin_amdgcn_exp2f(x);
#else
    return exp2f(x);
#endif
}

// 16-lane all-reduce max (one WMMA row). DPP16 butterfly: pure VALU, no DS.
__device__ __forceinline__ float wave16_max(float x)
{
#if __has_builtin(__builtin_amdgcn_update_dpp)
    union { float f; int i; } u, v;
    u.f = x;
    v.i = __builtin_amdgcn_update_dpp(0, u.i, 0xB1, 0xf, 0xf, true);   // quad_perm(1,0,3,2)
    x = fmaxf(x, v.f); u.f = x;
    v.i = __builtin_amdgcn_update_dpp(0, u.i, 0x4E, 0xf, 0xf, true);   // quad_perm(2,3,0,1)
    x = fmaxf(x, v.f); u.f = x;
    v.i = __builtin_amdgcn_update_dpp(0, u.i, 0x124, 0xf, 0xf, true);  // row_ror:4
    x = fmaxf(x, v.f); u.f = x;
    v.i = __builtin_amdgcn_update_dpp(0, u.i, 0x128, 0xf, 0xf, true);  // row_ror:8
    x = fmaxf(x, v.f);
    return x;
#else
    x = fmaxf(x, __shfl_xor(x, 1));
    x = fmaxf(x, __shfl_xor(x, 2));
    x = fmaxf(x, __shfl_xor(x, 4));
    x = fmaxf(x, __shfl_xor(x, 8));
    return x;
#endif
}

// ---------------------------------------------------------------------------
// fp32 -> bf16 elementwise conversion (8 elems / thread).
// ---------------------------------------------------------------------------
__global__ __launch_bounds__(256)
void cvt_bf16_kernel(const float* __restrict__ in, __bf16* __restrict__ out, int n)
{
    const int i = (blockIdx.x * 256 + threadIdx.x) * 8;
    if (i >= n) return;
    const f32x4 a = *(const f32x4*)(in + i);
    const f32x4 b = *(const f32x4*)(in + i + 4);
    bf16x8 o;
#pragma unroll
    for (int j = 0; j < 4; ++j) { o[j] = (__bf16)a[j]; o[4 + j] = (__bf16)b[j]; }
    *(bf16x8*)(out + i) = o;
}

// ---------------------------------------------------------------------------
// GEMM: C[m,n] = (sum_k X[m,k]*W[n,k] + bias[n]) * outScale
//   X, W bf16. M=8192, N=K=1024. Block tile 128x128, K-step 32, 8 waves,
//   wave tile 32x64. Double-buffered TDM staging with padded 48-elem stride.
//   MODE 0: bf16 -> [b,h,s,dk]   MODE 1: bf16 -> [b,h,dk,s]   MODE 2: f32 [m,n]
// ---------------------------------------------------------------------------
template <int MODE>
__global__ __launch_bounds__(256)
void gemm_wmma_kernel(const __bf16* __restrict__ X, const __bf16* __restrict__ W,
                      const float* __restrict__ bias, void* __restrict__ Out,
                      float outScale)
{
    constexpr int K = D_MODEL;
    constexpr int LDSW = 48;                           // padded bf16 row stride
    constexpr unsigned TILEB = 128 * LDSW * 2;         // 12 KB per buffer
    __shared__ __align__(32) __bf16 As[2][128 * LDSW];
    __shared__ __align__(32) __bf16 Bs[2][128 * LDSW];

    const int tid   = threadIdx.x;
    const int wave  = tid >> 5;
    const int lane  = tid & 31;
    const int lrow  = lane & 15;
    const int lhalf = lane >> 4;

    const int mBlock = blockIdx.y * 128;
    const int nBlock = blockIdx.x * 128;
    const int wm = wave >> 1;
    const int wn = wave & 1;

    f32x8 acc[2][4];
#pragma unroll
    for (int mt = 0; mt < 2; ++mt)
#pragma unroll
        for (int nt = 0; nt < 4; ++nt) acc[mt][nt] = {};

    const unsigned asAddr = (unsigned)(uintptr_t)(void*)&As[0][0];
    const unsigned bsAddr = (unsigned)(uintptr_t)(void*)&Bs[0][0];
    constexpr int NIT = K / 32;

#if defined(HAVE_TDM)
    if (wave == 0) {       // prologue: tile 0 into buffer 0
        tdm_load_2d(asAddr, X + (size_t)mBlock * K, K, 128, 32, 128, K, true);
        tdm_load_2d(bsAddr, W + (size_t)nBlock * K, K, 128, 32, 128, K, true);
    }
#endif
    for (int it = 0; it < NIT; ++it) {
        const int cur = it & 1;
        const int k0  = it * 32;
#if defined(HAVE_TDM)
        if (wave == 0) {
            if (it + 1 < NIT) {   // overlap: issue next tile, then wait for current
                const unsigned nb = (unsigned)((it + 1) & 1) * TILEB;
                tdm_load_2d(asAddr + nb, X + (size_t)mBlock * K + k0 + 32, K, 128, 32, 128, K, true);
                tdm_load_2d(bsAddr + nb, W + (size_t)nBlock * K + k0 + 32, K, 128, 32, 128, K, true);
                __builtin_amdgcn_s_wait_tensorcnt(2);
            } else {
                __builtin_amdgcn_s_wait_tensorcnt(0);
            }
        }
#else
        {   // cooperative fallback
            const int r = tid >> 1, c = (tid & 1) * 16;
            *(bf16x8*)&As[cur][r * LDSW + c]     = *(const bf16x8*)&X[(size_t)(mBlock + r) * K + k0 + c];
            *(bf16x8*)&As[cur][r * LDSW + c + 8] = *(const bf16x8*)&X[(size_t)(mBlock + r) * K + k0 + c + 8];
            *(bf16x8*)&Bs[cur][r * LDSW + c]     = *(const bf16x8*)&W[(size_t)(nBlock + r) * K + k0 + c];
            *(bf16x8*)&Bs[cur][r * LDSW + c + 8] = *(const bf16x8*)&W[(size_t)(nBlock + r) * K + k0 + c + 8];
        }
#endif
        __syncthreads();

        bf16x16 afrag[2], bfrag[4];
#pragma unroll
        for (int mt = 0; mt < 2; ++mt)
            afrag[mt] = *(const bf16x16*)&As[cur][(wm * 32 + mt * 16 + lrow) * LDSW + lhalf * 16];
#pragma unroll
        for (int nt = 0; nt < 4; ++nt)
            bfrag[nt] = *(const bf16x16*)&Bs[cur][(wn * 64 + nt * 16 + lrow) * LDSW + lhalf * 16];
#pragma unroll
        for (int mt = 0; mt < 2; ++mt)
#pragma unroll
            for (int nt = 0; nt < 4; ++nt)
                acc[mt][nt] = __builtin_amdgcn_wmma_f32_16x16x32_bf16(
                    false, afrag[mt], false, bfrag[nt],
                    (short)0, acc[mt][nt], false, false);
        __syncthreads();   // protect the buffer the next iteration's TDM overwrites
    }

#pragma unroll
    for (int nt = 0; nt < 4; ++nt) {
        const int n = nBlock + wn * 64 + nt * 16 + lrow;
        const float bv = bias[n];
#pragma unroll
        for (int mt = 0; mt < 2; ++mt) {
#pragma unroll
            for (int r = 0; r < 8; ++r) {
                const int m = mBlock + wm * 32 + mt * 16 + r + 8 * lhalf;
                const float v = (acc[mt][nt][r] + bv) * outScale;
                if (MODE == 0) {
                    const int b = m >> 11, s = m & (SEQ - 1);
                    const int h = n >> 6,  dk = n & (DK - 1);
                    ((__bf16*)Out)[(((size_t)(b * NHEADS + h) * SEQ + s) << 6) + dk] = (__bf16)v;
                } else if (MODE == 1) {
                    const int b = m >> 11, s = m & (SEQ - 1);
                    const int h = n >> 6,  dk = n & (DK - 1);
                    ((__bf16*)Out)[((size_t)((b * NHEADS + h) * DK + dk) << 11) + s] = (__bf16)v;
                } else {
                    ((float*)Out)[(size_t)m * D_MODEL + n] = v;
                }
            }
        }
    }
}

// ---------------------------------------------------------------------------
// Flash attention: block = (b,h) x 128 q rows; wave = 16-row q tile.
// 64-key blocks double-buffered into LDS by the TDM. Softmax in base-2
// domain (log2e/sqrt(dk) folded into Q projection); row-max via DPP16,
// row-sum via WMMA against an all-ones B fragment.
// ---------------------------------------------------------------------------
__global__ __launch_bounds__(256)
void flash_attn_wmma_kernel(const __bf16* __restrict__ Qh,
                            const __bf16* __restrict__ Kh,
                            const __bf16* __restrict__ VhT,
                            __bf16* __restrict__ CtxOut)
{
    constexpr unsigned KT_B = 64 * DK * 2;             // 8 KB per buffer
    __shared__ __align__(32) __bf16 Kt[2][64 * DK];    // [key][dk]
    __shared__ __align__(32) __bf16 Vt[2][DK * 64];    // [dk][key]
    __shared__ __align__(32) __bf16 Plds[8][16][64];   // per-wave P

    const int wave  = threadIdx.x >> 5;
    const int lane  = threadIdx.x & 31;
    const int lrow  = lane & 15;
    const int lhalf = lane >> 4;

    const int bh = blockIdx.y;
    const int q0 = blockIdx.x * 128 + wave * 16;

    const __bf16* Qbase = Qh  + (size_t)bh * SEQ * DK;
    const __bf16* Kbase = Kh  + (size_t)bh * SEQ * DK;
    const __bf16* Vbase = VhT + (size_t)bh * DK * SEQ;

    bf16x16 aq[2];
#pragma unroll
    for (int c = 0; c < 2; ++c)
        aq[c] = *(const bf16x16*)&Qbase[(size_t)(q0 + lrow) * DK + c * 32 + lhalf * 16];

    bf16x16 onesB;
#pragma unroll
    for (int i = 0; i < 16; ++i) onesB[i] = (__bf16)1.0f;

    f32x8 acc[4], lacc = {};
#pragma unroll
    for (int d = 0; d < 4; ++d) acc[d] = {};
    float mmax[8];
#pragma unroll
    for (int r = 0; r < 8; ++r) mmax[r] = -3.0e38f;

    const unsigned ktAddr = (unsigned)(uintptr_t)(void*)&Kt[0][0];
    const unsigned vtAddr = (unsigned)(uintptr_t)(void*)&Vt[0][0];
    constexpr int NIT = SEQ / 64;

#if defined(HAVE_TDM)
    if (wave == 0) {       // prologue: block 0 into buffer 0
        tdm_load_2d(ktAddr, Kbase, DK, 64, DK, 64, DK, false);
        tdm_load_2d(vtAddr, Vbase, SEQ, DK, 64, DK, SEQ, false);
    }
#endif
    for (int it = 0; it < NIT; ++it) {
        const int cur = it & 1;
        const int j0  = it * 64;
#if defined(HAVE_TDM)
        if (wave == 0) {
            if (it + 1 < NIT) {
                const unsigned nb = (unsigned)((it + 1) & 1) * KT_B;
                tdm_load_2d(ktAddr + nb, Kbase + (size_t)(j0 + 64) * DK, DK, 64, DK, 64, DK, false);
                tdm_load_2d(vtAddr + nb, Vbase + j0 + 64, SEQ, DK, 64, DK, SEQ, false);
                __builtin_amdgcn_s_wait_tensorcnt(2);
            } else {
                __builtin_amdgcn_s_wait_tensorcnt(0);
            }
        }
#else
        {   // cooperative fallback
            const int r = threadIdx.x >> 2, c = (threadIdx.x & 3) * 16;
            *(bf16x8*)&Kt[cur][r * DK + c]     = *(const bf16x8*)&Kbase[(size_t)(j0 + r) * DK + c];
            *(bf16x8*)&Kt[cur][r * DK + c + 8] = *(const bf16x8*)&Kbase[(size_t)(j0 + r) * DK + c + 8];
            *(bf16x8*)&Vt[cur][r * 64 + c]     = *(const bf16x8*)&Vbase[(size_t)r * SEQ + j0 + c];
            *(bf16x8*)&Vt[cur][r * 64 + c + 8] = *(const bf16x8*)&Vbase[(size_t)r * SEQ + j0 + c + 8];
        }
#endif
        __syncthreads();

        // ---- scores: 16 q x 64 keys (log2 domain) ----
        f32x8 sc[4];
#pragma unroll
        for (int nt = 0; nt < 4; ++nt) {
            sc[nt] = {};
#pragma unroll
            for (int c = 0; c < 2; ++c) {
                const bf16x16 bk = *(const bf16x16*)&Kt[cur][(nt * 16 + lrow) * DK + c * 32 + lhalf * 16];
                sc[nt] = __builtin_amdgcn_wmma_f32_16x16x32_bf16(
                    false, aq[c], false, bk, (short)0, sc[nt], false, false);
            }
        }
        // ---- online softmax: DPP16 row-max, rescale, P -> LDS ----
#pragma unroll
        for (int r = 0; r < 8; ++r) {
            float rm = fmaxf(fmaxf(sc[0][r], sc[1][r]), fmaxf(sc[2][r], sc[3][r]));
            rm = wave16_max(rm);
            const float nm   = fmaxf(mmax[r], rm);
            const float corr = fast_exp2(mmax[r] - nm);
            mmax[r] = nm;
            const int prow = r + 8 * lhalf;
#pragma unroll
            for (int nt = 0; nt < 4; ++nt) {
                const float p = fast_exp2(sc[nt][r] - nm);
                Plds[wave][prow][nt * 16 + lrow] = (__bf16)p;
            }
            lacc[r] *= corr;
#pragma unroll
            for (int d = 0; d < 4; ++d) acc[d][r] *= corr;
        }
        asm volatile("s_wait_dscnt 0" ::: "memory");   // same-wave P RAW
        bf16x16 ap[2];
#pragma unroll
        for (int c = 0; c < 2; ++c)
            ap[c] = *(const bf16x16*)&Plds[wave][lrow][c * 32 + lhalf * 16];
        // ---- row-sum of P via WMMA against all-ones B ----
#pragma unroll
        for (int c = 0; c < 2; ++c)
            lacc = __builtin_amdgcn_wmma_f32_16x16x32_bf16(
                false, ap[c], false, onesB, (short)0, lacc, false, false);
        // ---- ctx += P (16x64) @ V (64x64) ----
#pragma unroll
        for (int d = 0; d < 4; ++d)
#pragma unroll
            for (int c = 0; c < 2; ++c) {
                const bf16x16 bv = *(const bf16x16*)&Vt[cur][(d * 16 + lrow) * 64 + c * 32 + lhalf * 16];
                acc[d] = __builtin_amdgcn_wmma_f32_16x16x32_bf16(
                    false, ap[c], false, bv, (short)0, acc[d], false, false);
            }
        __syncthreads();   // protect Kt/Vt buffer against next TDM write
    }

    const int b = bh >> 4, h = bh & (NHEADS - 1);
#pragma unroll
    for (int r = 0; r < 8; ++r) {
        const float inv  = 1.0f / lacc[r];
        const int   srow = q0 + r + 8 * lhalf;
        __bf16* dst = CtxOut + ((size_t)(b * SEQ + srow) * D_MODEL) + h * DK;
#pragma unroll
        for (int d = 0; d < 4; ++d)
            dst[d * 16 + lrow] = (__bf16)(acc[d][r] * inv);
    }
}

// ---------------------------------------------------------------------------
// Host launcher.
// ws: Qb Kb Vb (bf16, 16MB ea) | Wq..Wo (bf16, 2MB ea) | qh kh vhT ctx (16MB ea)
// ---------------------------------------------------------------------------
extern "C" void kernel_launch(void* const* d_in, const int* in_sizes, int n_in,
                              void* d_out, int out_size, void* d_ws, size_t ws_size,
                              hipStream_t stream)
{
    (void)in_sizes; (void)n_in; (void)out_size; (void)ws_size;
    const float* Q  = (const float*)d_in[0];
    const float* K  = (const float*)d_in[1];
    const float* V  = (const float*)d_in[2];
    const float* Wq = (const float*)d_in[3];
    const float* bq = (const float*)d_in[4];
    const float* Wk = (const float*)d_in[5];
    const float* bk = (const float*)d_in[6];
    const float* Wv = (const float*)d_in[7];
    const float* bv = (const float*)d_in[8];
    const float* Wo = (const float*)d_in[9];
    const float* bo = (const float*)d_in[10];

    char* ws = (char*)d_ws;
    const size_t NX = (size_t)BATCH * SEQ * D_MODEL;   // 8M elements
    const size_t NW = (size_t)D_MODEL * D_MODEL;       // 1M elements
    __bf16* Qb  = (__bf16*)(ws);
    __bf16* Kb  = (__bf16*)(ws + 2 * NX);
    __bf16* Vb  = (__bf16*)(ws + 4 * NX);
    __bf16* Wqb = (__bf16*)(ws + 6 * NX);
    __bf16* Wkb = (__bf16*)(ws + 6 * NX + 2 * NW);
    __bf16* Wvb = (__bf16*)(ws + 6 * NX + 4 * NW);
    __bf16* Wob = (__bf16*)(ws + 6 * NX + 6 * NW);
    __bf16* qh  = (__bf16*)(ws + 6 * NX + 8 * NW);
    __bf16* kh  = (__bf16*)(ws + 8 * NX + 8 * NW);
    __bf16* vht = (__bf16*)(ws + 10 * NX + 8 * NW);
    __bf16* ctx = (__bf16*)(ws + 12 * NX + 8 * NW);

    const int cvtBlkX = (int)(NX / (256 * 8));
    const int cvtBlkW = (int)(NW / (256 * 8));
    cvt_bf16_kernel<<<cvtBlkX, 256, 0, stream>>>(Q,  Qb,  (int)NX);
    cvt_bf16_kernel<<<cvtBlkX, 256, 0, stream>>>(K,  Kb,  (int)NX);
    cvt_bf16_kernel<<<cvtBlkX, 256, 0, stream>>>(V,  Vb,  (int)NX);
    cvt_bf16_kernel<<<cvtBlkW, 256, 0, stream>>>(Wq, Wqb, (int)NW);
    cvt_bf16_kernel<<<cvtBlkW, 256, 0, stream>>>(Wk, Wkb, (int)NW);
    cvt_bf16_kernel<<<cvtBlkW, 256, 0, stream>>>(Wv, Wvb, (int)NW);
    cvt_bf16_kernel<<<cvtBlkW, 256, 0, stream>>>(Wo, Wob, (int)NW);

    const dim3 ggrid(D_MODEL / 128, (BATCH * SEQ) / 128);   // (8, 64)
    const dim3 gblk(256);
    const float qScale = 0.125f * 1.44269504088896340736f;  // 1/sqrt(dk) * log2(e)

    gemm_wmma_kernel<0><<<ggrid, gblk, 0, stream>>>(Qb, Wqb, bq, (void*)qh, qScale);
    gemm_wmma_kernel<0><<<ggrid, gblk, 0, stream>>>(Kb, Wkb, bk, (void*)kh, 1.0f);
    gemm_wmma_kernel<1><<<ggrid, gblk, 0, stream>>>(Vb, Wvb, bv, (void*)vht, 1.0f);

    flash_attn_wmma_kernel<<<dim3(SEQ / 128, BATCH * NHEADS), gblk, 0, stream>>>(
        qh, kh, vht, ctx);

    gemm_wmma_kernel<2><<<ggrid, gblk, 0, stream>>>(ctx, Wob, bo, d_out, 1.0f);
}
